// RecurrentRetention_34660386079008
// MI455X (gfx1250) — matrix operand order repaired and checked by reading, not compiled, not run on hardware
//
#include <hip/hip_runtime.h>
#include <hip/hip_bf16.h>

// ---------------------------------------------------------------------------
// RetNet recurrent retention for MI455X (gfx1250, wave32).
//   b=2, n=16, l=1024, e=512, qd=kd=vd=64
// Bandwidth-bound by the 537MB current_kv write (NT stores, since it far
// exceeds the 192MB L2 and is never re-read). Projections run through
// v_wmma_f32_16x16x32_bf16; past_kv staging uses global_load_async_to_lds.
// ---------------------------------------------------------------------------

typedef __attribute__((ext_vector_type(16))) __bf16 v16bf;
typedef __attribute__((ext_vector_type(8)))  __bf16 v8bf;
typedef __attribute__((ext_vector_type(8)))  float  v8f;
typedef __attribute__((ext_vector_type(2)))  float  v2f;

#define BB     2
#define NH     16
#define LL     1024
#define EE     512
#define DD     64
#define HEADS  (BB * NH)        // 32
#define M_TOT  (HEADS * LL)     // 32768 rows for the projection GEMM

// ---------------------------------------------------------------------------
// Kernel 1: convert Wq/Wk/Wv (fp32 [64,512]) to bf16, stacked [3,64,512].
// ---------------------------------------------------------------------------
__global__ void retnet_wconv(const float* __restrict__ Wq,
                             const float* __restrict__ Wk,
                             const float* __restrict__ Wv,
                             __bf16* __restrict__ wbf) {
    int i = blockIdx.x * blockDim.x + threadIdx.x;
    if (i >= 3 * DD * EE) return;
    const float* src = (i < DD * EE) ? Wq : ((i < 2 * DD * EE) ? Wk : Wv);
    wbf[i] = (__bf16)src[i % (DD * EE)];
}

// ---------------------------------------------------------------------------
// Kernel 2: QKV projection.  out[m, d] = sum_e x[m, e] * W[d, e] + bias[d]
// One block = 128 threads (4 waves) = one 16-row tile of x, full 64 output
// columns of one projection (grid.z selects q/k/v).  x tile is staged in LDS
// as bf16 in contiguous k = 16*(lane/16) + s order, as B; each wave does a 16x16
// output tile via v_wmma_f32_16x16x32_bf16 over K=512 in steps of 32.
// ---------------------------------------------------------------------------
__global__ __launch_bounds__(128)
void retnet_proj(const float* __restrict__ x,
                 const __bf16* __restrict__ wbf,
                 const float* __restrict__ bq,
                 const float* __restrict__ bk,
                 const float* __restrict__ bv,
                 float* __restrict__ qb,
                 float* __restrict__ kb,
                 float* __restrict__ vb) {
    __shared__ __align__(32) __bf16 lds_a[32 * 16];   // 1 KB A fragment

    const int tid  = threadIdx.x;
    const int lane = tid & 31;
    const int wave = tid >> 5;
    const int z    = blockIdx.z;
    const int row0 = blockIdx.x * 16;

    const __bf16* wz   = wbf + (size_t)z * (DD * EE);
    const float*  bias = (z == 0) ? bq : ((z == 1) ? bk : bv);
    float*        dst  = (z == 0) ? qb : ((z == 1) ? kb : vb);

    const int ncol  = wave * 16 + (lane & 15);
    const int khalf = lane >> 4;

    // Per-lane A-staging geometry (used when tid < 64).
    const int laneA = tid >> 1;
    const int half  = tid & 1;
    const float* asrc =
        x + (size_t)(row0 + (laneA & 15)) * EE + ((laneA >> 4) * 16) + half * 8;

    v8f acc = {};

    for (int kc = 0; kc < EE; kc += 32) {
        // --- stage x tile (16 rows x 32 K) into LDS as bf16, A-frag order ---
        if (tid < 64) {
            float4 a0 = *(const float4*)(asrc + kc);
            float4 a1 = *(const float4*)(asrc + kc + 4);
            if (kc + 32 < EE)
                __builtin_prefetch(asrc + kc + 32, 0, 0);
            v8bf t;
            t[0] = (__bf16)a0.x; t[1] = (__bf16)a0.y;
            t[2] = (__bf16)a0.z; t[3] = (__bf16)a0.w;
            t[4] = (__bf16)a1.x; t[5] = (__bf16)a1.y;
            t[6] = (__bf16)a1.z; t[7] = (__bf16)a1.w;
            *(v8bf*)(lds_a + laneA * 16 + half * 8) = t;
        }
        __syncthreads();

        v16bf afrag = *(const v16bf*)(lds_a + lane * 16);
        v16bf bfrag =
            *(const v16bf*)(wz + (size_t)ncol * EE + kc + khalf * 16);

        // D = A x B + C   (8 args: neg_a, A, neg_b, B, c_mod, C, reuse_a, reuse_b)
        acc = __builtin_amdgcn_wmma_f32_16x16x32_bf16(
            false, afrag, false, bfrag, (short)0, acc, false, false);

        __syncthreads();
    }

    // C/D layout: element r of lane => row = row0 + 8*(lane>>4) + r, col = ncol
    const float biasv  = bias[ncol];
    const int rowbase = row0 + khalf * 8;
    #pragma unroll
    for (int r = 0; r < 8; ++r) {
        dst[(size_t)(rowbase + r) * DD + ncol] = acc[r] + biasv;
    }
}

// ---------------------------------------------------------------------------
// Kernel 3: retention core (streaming phase, HBM-write bound).
// Per position l:  ckv[l] = decay*past_kv + k_l (x) v_l
//                  out[l] = decay*(q_l @ past_kv) + (q_l . k_l) * v_l
// Block = 256 thr (8 waves) per (head, 128-position chunk). past_kv is copied
// into LDS with global_load_async_to_lds_b128 (ASYNCcnt), scaled by decay in
// place; each wave handles one position per iteration, lane owns 2 v-columns
// => ckv row stores are coalesced 256B non-temporal stores per wave (ckv is
// 537MB, write-once: keep it out of the 192MB L2).  q/k/v are read once =>
// non-temporal loads.  (q,k) pairs staged in per-wave LDS so broadcast reads
// are conflict-free.  Per-block (sum,sumsq) partials for group-norm.
// ---------------------------------------------------------------------------
__global__ __launch_bounds__(256)
void retnet_core(const float* __restrict__ qb,
                 const float* __restrict__ kb,
                 const float* __restrict__ vb,
                 const float* __restrict__ past_kv,
                 float* __restrict__ out,
                 float* __restrict__ ckv,
                 float* __restrict__ partials) {
    __shared__ __align__(16) float lds_dpkv[DD * DD];   // 16 KB
    __shared__ float lds_qk[8 * DD * 2];                // 4 KB per-wave pairs
    __shared__ float redS[256], redS2[256];

    const int tid  = threadIdx.x;
    const int lane = tid & 31;
    const int wave = tid >> 5;
    const int h    = blockIdx.x;               // 0..31  (b*16 + n)
    const int ly   = blockIdx.y;               // 0..7   (128-position chunk)
    const int n    = h & (NH - 1);
    const float decay = 1.0f - exp2f(-5.0f - (float)n);

    // --- async copy past_kv (16 KB) into LDS: 4 x b128 per thread ----------
    {
        const float* gbase = past_kv + (size_t)h * DD * DD;
        const unsigned lds_base = (unsigned)(uintptr_t)(void*)lds_dpkv;
        #pragma unroll
        for (int j = 0; j < 4; ++j) {
            unsigned boff   = (unsigned)(tid + j * 256) * 16u;  // byte offset
            unsigned ldsoff = lds_base + boff;
            asm volatile(
                "global_load_async_to_lds_b128 %0, %1, %2 offset:0"
                :
                : "v"(ldsoff), "v"(boff), "s"(gbase)
                : "memory");
        }
        asm volatile("s_wait_asynccnt 0x0" ::: "memory");
        // scale own chunks by decay in place (no cross-thread deps yet)
        float4* dp4 = (float4*)lds_dpkv;
        #pragma unroll
        for (int j = 0; j < 4; ++j) {
            int id = tid + j * 256;
            float4 p = dp4[id];
            dp4[id] = make_float4(decay * p.x, decay * p.y,
                                  decay * p.z, decay * p.w);
        }
    }
    __syncthreads();

    const v2f* dpkv2 = (const v2f*)lds_dpkv;            // [kk*32 + lane]
    v2f* qkW = (v2f*)lds_qk + wave * DD;                // wave-private

    float s = 0.0f, s2 = 0.0f;

    for (int it = 0; it < 16; ++it) {
        const int l = ly * 128 + it * 8 + wave;
        const size_t m = (size_t)h * LL + l;

        v2f q2 = __builtin_nontemporal_load((const v2f*)(qb + m * DD) + lane);
        v2f k2 = __builtin_nontemporal_load((const v2f*)(kb + m * DD) + lane);
        v2f v2 = __builtin_nontemporal_load((const v2f*)(vb + m * DD) + lane);

        if (it < 15) {   // prefetch next position's rows (global_prefetch_b8)
            __builtin_prefetch(qb + (m + 8) * DD + 2 * lane, 0, 0);
            __builtin_prefetch(kb + (m + 8) * DD + 2 * lane, 0, 0);
            __builtin_prefetch(vb + (m + 8) * DD + 2 * lane, 0, 0);
        }

        // stage interleaved (q,k) pairs: qkW[2*lane]=(q.x,k.x), [2*lane+1]=(q.y,k.y)
        *(float4*)(qkW + 2 * lane) = make_float4(q2.x, k2.x, q2.y, k2.y);

        v2f acc = {0.0f, 0.0f};
        float qkdot = 0.0f;
        float* cvbase = ckv + m * (size_t)(DD * DD);

        #pragma unroll 8
        for (int kk = 0; kk < DD; ++kk) {
            v2f qk = qkW[kk];                        // LDS broadcast
            v2f dp = dpkv2[kk * 32 + lane];          // conflict-free
            acc.x  = fmaf(qk.x, dp.x, acc.x);
            acc.y  = fmaf(qk.x, dp.y, acc.y);
            qkdot  = fmaf(qk.x, qk.y, qkdot);
            v2f cv;
            cv.x = fmaf(qk.y, v2.x, dp.x);
            cv.y = fmaf(qk.y, v2.y, dp.y);
            // coalesced 256B/wave, TH=NT: write-once 537MB stream
            __builtin_nontemporal_store(cv, (v2f*)(cvbase + kk * DD) + lane);
        }

        v2f o;
        o.x = fmaf(qkdot, v2.x, acc.x);
        o.y = fmaf(qkdot, v2.y, acc.y);
        ((v2f*)(out + m * DD))[lane] = o;            // re-read by gnorm: RT

        s  += o.x + o.y;
        s2 += o.x * o.x + o.y * o.y;
    }

    // Block reduction of (sum, sumsq) -> fixed partial slot (deterministic).
    redS[tid] = s; redS2[tid] = s2;
    __syncthreads();
    for (int st = 128; st > 0; st >>= 1) {
        if (tid < st) { redS[tid] += redS[tid + st]; redS2[tid] += redS2[tid + st]; }
        __syncthreads();
    }
    if (tid == 0) {
        partials[(h * 8 + ly) * 2 + 0] = redS[0];
        partials[(h * 8 + ly) * 2 + 1] = redS2[0];
    }
}

// ---------------------------------------------------------------------------
// Kernel 4: per-head mean / rstd from the 8 block partials.
// ---------------------------------------------------------------------------
__global__ void retnet_finalize(const float* __restrict__ partials,
                                float* __restrict__ meanrstd) {
    int h = threadIdx.x;
    if (h >= HEADS) return;
    float s = 0.0f, s2 = 0.0f;
    #pragma unroll
    for (int j = 0; j < 8; ++j) {
        s  += partials[(h * 8 + j) * 2 + 0];
        s2 += partials[(h * 8 + j) * 2 + 1];
    }
    const float inv = 1.0f / (float)(LL * DD);
    float mean = s * inv;
    float var  = s2 * inv - mean * mean;
    meanrstd[h * 2 + 0] = mean;
    meanrstd[h * 2 + 1] = rsqrtf(var + 1e-5f);
}

// ---------------------------------------------------------------------------
// Kernel 5: in-place group norm on the out region of d_out.
// ---------------------------------------------------------------------------
__global__ void retnet_gnorm(float* __restrict__ out,
                             const float* __restrict__ meanrstd) {
    int idx = blockIdx.x * blockDim.x + threadIdx.x;   // < 32*1024*64
    int h = idx >> 16;                                 // L*D = 65536 per head
    float mean = meanrstd[h * 2 + 0];
    float rstd = meanrstd[h * 2 + 1];
    out[idx] = (out[idx] - mean) * rstd;
}

// ---------------------------------------------------------------------------
extern "C" void kernel_launch(void* const* d_in, const int* in_sizes, int n_in,
                              void* d_out, int out_size, void* d_ws, size_t ws_size,
                              hipStream_t stream) {
    const float* x       = (const float*)d_in[0];
    const float* past_kv = (const float*)d_in[1];
    const float* Wq      = (const float*)d_in[2];
    const float* bq      = (const float*)d_in[3];
    const float* Wk      = (const float*)d_in[4];
    const float* bk      = (const float*)d_in[5];
    const float* Wv      = (const float*)d_in[6];
    const float* bv      = (const float*)d_in[7];

    float* out = (float*)d_out;                         // [2,16,1024,64]
    float* ckv = out + (size_t)HEADS * LL * DD;         // [2,16,1024,64,64]

    // Workspace layout (bytes):
    //   [0, 196608)        : bf16 weights, stacked [3,64,512]
    //   then q, k, v       : 3 x 32768x64 fp32 (8 MB each)
    //   then partials      : 32*8*2 fp32
    //   then mean/rstd     : 32*2 fp32
    char*   ws  = (char*)d_ws;
    __bf16* wbf = (__bf16*)ws;
    float*  qb  = (float*)(ws + 196608);
    float*  kb  = qb + (size_t)M_TOT * DD;
    float*  vb  = kb + (size_t)M_TOT * DD;
    float*  partials = vb + (size_t)M_TOT * DD;
    float*  meanrstd = partials + HEADS * 8 * 2;

    // 1) weight conversion: 3*64*512 = 98304 elements
    retnet_wconv<<<(3 * DD * EE + 255) / 256, 256, 0, stream>>>(Wq, Wk, Wv, wbf);

    // 2) QKV projections: grid.x = 32768/16 row tiles, grid.z selects q/k/v
    retnet_proj<<<dim3(M_TOT / 16, 1, 3), 128, 0, stream>>>(
        x, wbf, bq, bk, bv, qb, kb, vb);

    // 3) retention core: (head, 128-position chunk)
    retnet_core<<<dim3(HEADS, LL / 128), 256, 0, stream>>>(
        qb, kb, vb, past_kv, out, ckv, partials);

    // 4) per-head stats
    retnet_finalize<<<1, 32, 0, stream>>>(partials, meanrstd);

    // 5) in-place group norm on out
    retnet_gnorm<<<(HEADS * LL * DD) / 256, 256, 0, stream>>>(out, meanrstd);
}
